// W4A16LlamaMLP_83416854823015
// MI455X (gfx1250) — compile-verified, matrix operand order checked by reading, not run on hardware
//
#include <hip/hip_runtime.h>
#include <hip/hip_bf16.h>
#include <hip/hip_fp16.h>

// ---------------------------------------------------------------------------
// W4A16 Llama MLP for MI455X (gfx1250, wave32, WMMA), split-K edition.
//
// Roofline: 541 MB of int32-held int4 codes @ 23.3 TB/s -> ~23 us floor; the
// 17.3 GFLOP ride on v_wmma_f32_16x16x32_f16. Split-K (partials + fixed-order
// reduce) raises the grid from 86/32 blocks to ~688/256 so the whole chip
// streams weights concurrently. Dequant is done in packed f16 FMA form
// (q*s + (-z*s)) to cut VALU pressure vs f32 affine + downconvert.
// ---------------------------------------------------------------------------

typedef __attribute__((ext_vector_type(16))) _Float16 v16h;
typedef __attribute__((ext_vector_type(8)))  _Float16 v8h;
typedef __attribute__((ext_vector_type(8)))  float    v8f;

#define H_DIM     4096
#define I_DIM     11008
#define B_DIM     64
#define QGROUP    128
#define NGROUPS_H (H_DIM / QGROUP)   // 32
#define NGROUPS_I (I_DIM / QGROUP)   // 86

// ------------------------- fp32 -> fp16 activation cast ---------------------
__global__ void cvt_x_f16(const float* __restrict__ x, _Float16* __restrict__ xh, int n) {
    int idx = blockIdx.x * blockDim.x + threadIdx.x;
    if (idx < n) xh[idx] = (_Float16)x[idx];
}

// Dequantize 16 int4 codes (held as int32) into a WMMA B-fragment (v16h)
// using f16 math: w = (f16)q * scale_h + bias_h, bias = -zero*scale.
// Fragment layout (16-bit): elems 0..7 -> K = base0..base0+7, elems 8..15 ->
// K = 16+base0..16+base0+7 (base0 = 0 lanes 0-15, 8 lanes 16-31).
__device__ __forceinline__ v16h dequant_frag(const int* __restrict__ qrow,
                                             int kb, int base0,
                                             _Float16 sh, _Float16 bh) {
    const int4* p0 = (const int4*)(qrow + kb + base0);
    const int4* p1 = (const int4*)(qrow + kb + 16 + base0);
    int4 q0 = p0[0], q1 = p0[1];
    int4 q2 = p1[0], q3 = p1[1];
    v16h r;
    r[0]  = (_Float16)(short)q0.x * sh + bh;
    r[1]  = (_Float16)(short)q0.y * sh + bh;
    r[2]  = (_Float16)(short)q0.z * sh + bh;
    r[3]  = (_Float16)(short)q0.w * sh + bh;
    r[4]  = (_Float16)(short)q1.x * sh + bh;
    r[5]  = (_Float16)(short)q1.y * sh + bh;
    r[6]  = (_Float16)(short)q1.z * sh + bh;
    r[7]  = (_Float16)(short)q1.w * sh + bh;
    r[8]  = (_Float16)(short)q2.x * sh + bh;
    r[9]  = (_Float16)(short)q2.y * sh + bh;
    r[10] = (_Float16)(short)q2.z * sh + bh;
    r[11] = (_Float16)(short)q2.w * sh + bh;
    r[12] = (_Float16)(short)q3.x * sh + bh;
    r[13] = (_Float16)(short)q3.y * sh + bh;
    r[14] = (_Float16)(short)q3.z * sh + bh;
    r[15] = (_Float16)(short)q3.w * sh + bh;
    return r;
}

// Load an f16 A-fragment: p already offset to (row, kb + base0); elements 0..7
// contiguous there, elements 8..15 contiguous at +16 halfs. Two b128 loads.
__device__ __forceinline__ v16h load_a_frag(const _Float16* __restrict__ p) {
    union { v16h v; v8h h[2]; } u;
    u.h[0] = *(const v8h*)(p);
    u.h[1] = *(const v8h*)(p + 16);
    return u.v;
}

__device__ __forceinline__ float silu(float g) {
    return g / (1.0f + __expf(-g));
}

// -------- kernel 1a: gate + up partial GEMM (split-K over H groups) ---------
// Grid: (I_DIM/128, KS1) x 256 threads (8 waves). Wave owns 16 I-columns,
// 4 M-tiles x {gate,up} accumulators. blockIdx.y selects the K-slice.
__global__ void __launch_bounds__(256)
gate_up_partial(const _Float16* __restrict__ xh,
                const int*   __restrict__ gq, const float* __restrict__ gs,
                const float* __restrict__ gz,
                const int*   __restrict__ uq, const float* __restrict__ us,
                const float* __restrict__ uz,
                float* __restrict__ pg, float* __restrict__ pu) {
    const int lane  = threadIdx.x & 31;
    const int wave  = threadIdx.x >> 5;
    const int col   = blockIdx.x * 128 + wave * 16 + (lane & 15); // I index
    const int base0 = (lane >> 4) * 8;
    const int mrow  = lane & 15;

    const int gpb = NGROUPS_H / gridDim.y;           // groups per split (exact)
    const int g0  = blockIdx.y * gpb;
    const int g1  = g0 + gpb;

    const int* __restrict__ gqrow = gq + (size_t)col * H_DIM;
    const int* __restrict__ uqrow = uq + (size_t)col * H_DIM;

    v8f vzero = {};
    v8f accg[4], accu[4];
#pragma unroll
    for (int m = 0; m < 4; ++m) { accg[m] = vzero; accu[m] = vzero; }

    for (int g = g0; g < g1; ++g) {
        if (g + 1 < g1) {
            __builtin_prefetch(gqrow + (g + 1) * QGROUP + base0, 0, 1);
            __builtin_prefetch(uqrow + (g + 1) * QGROUP + base0, 0, 1);
        }
        const float sg = gs[col * NGROUPS_H + g];
        const float zg = gz[col * NGROUPS_H + g];
        const float su = us[col * NGROUPS_H + g];
        const float zu = uz[col * NGROUPS_H + g];
        const _Float16 sgh = (_Float16)sg, bgh = (_Float16)(-zg * sg);
        const _Float16 suh = (_Float16)su, buh = (_Float16)(-zu * su);
#pragma unroll
        for (int t = 0; t < 4; ++t) {
            const int kb = g * QGROUP + t * 32;
            v16h bg = dequant_frag(gqrow, kb, base0, sgh, bgh);
            v16h bu = dequant_frag(uqrow, kb, base0, suh, buh);
#pragma unroll
            for (int m = 0; m < 4; ++m) {
                v16h a = load_a_frag(xh + (size_t)(m * 16 + mrow) * H_DIM + kb + base0);
                accg[m] = __builtin_amdgcn_wmma_f32_16x16x32_f16(
                    false, a, false, bg, (short)0, accg[m], false, false);
                accu[m] = __builtin_amdgcn_wmma_f32_16x16x32_f16(
                    false, a, false, bu, (short)0, accu[m], false, false);
            }
        }
    }

    // C/D layout: VGPR r -> M = r (lanes 0-15) or r+8 (lanes 16-31); N = lane&15.
    const int rowhi = (lane >> 4) * 8;
    const size_t sbase = (size_t)blockIdx.y * B_DIM * I_DIM;
#pragma unroll
    for (int m = 0; m < 4; ++m) {
#pragma unroll
        for (int r = 0; r < 8; ++r) {
            const int b = m * 16 + r + rowhi;
            pg[sbase + (size_t)b * I_DIM + col] = accg[m][r];
            pu[sbase + (size_t)b * I_DIM + col] = accu[m][r];
        }
    }
}

// -------- kernel 1b: reduce split-K partials + SwiGLU -> h (f16) ------------
__global__ void reduce_swiglu(const float* __restrict__ pg, const float* __restrict__ pu,
                              int ks, _Float16* __restrict__ hbuf, int n) {
    int idx = blockIdx.x * blockDim.x + threadIdx.x;
    if (idx >= n) return;
    float g = 0.0f, u = 0.0f;
    for (int s = 0; s < ks; ++s) {           // fixed order -> deterministic
        g += pg[(size_t)s * n + idx];
        u += pu[(size_t)s * n + idx];
    }
    hbuf[idx] = (_Float16)(silu(g) * u);
}

// -------- kernel 2a: down-proj partial GEMM (split-K over I groups) ---------
__global__ void __launch_bounds__(256)
down_partial(const _Float16* __restrict__ hbuf,
             const int*   __restrict__ dq, const float* __restrict__ ds,
             const float* __restrict__ dz,
             float* __restrict__ pd) {
    const int lane  = threadIdx.x & 31;
    const int wave  = threadIdx.x >> 5;
    const int col   = blockIdx.x * 128 + wave * 16 + (lane & 15); // H index
    const int base0 = (lane >> 4) * 8;
    const int mrow  = lane & 15;

    const int g0 = (int)(((long)blockIdx.y * NGROUPS_I) / gridDim.y);
    const int g1 = (int)(((long)(blockIdx.y + 1) * NGROUPS_I) / gridDim.y);

    const int* __restrict__ qrow = dq + (size_t)col * I_DIM;

    v8f vzero = {};
    v8f acc[4];
#pragma unroll
    for (int m = 0; m < 4; ++m) acc[m] = vzero;

    for (int g = g0; g < g1; ++g) {
        if (g + 1 < g1)
            __builtin_prefetch(qrow + (g + 1) * QGROUP + base0, 0, 1);
        const float s = ds[col * NGROUPS_I + g];
        const float z = dz[col * NGROUPS_I + g];
        const _Float16 sh = (_Float16)s, bh = (_Float16)(-z * s);
#pragma unroll
        for (int t = 0; t < 4; ++t) {
            const int kb = g * QGROUP + t * 32;
            v16h b = dequant_frag(qrow, kb, base0, sh, bh);
#pragma unroll
            for (int m = 0; m < 4; ++m) {
                v16h a = load_a_frag(hbuf + (size_t)(m * 16 + mrow) * I_DIM + kb + base0);
                acc[m] = __builtin_amdgcn_wmma_f32_16x16x32_f16(
                    false, a, false, b, (short)0, acc[m], false, false);
            }
        }
    }

    const int rowhi = (lane >> 4) * 8;
    const size_t sbase = (size_t)blockIdx.y * B_DIM * H_DIM;
#pragma unroll
    for (int m = 0; m < 4; ++m) {
#pragma unroll
        for (int r = 0; r < 8; ++r) {
            const int b = m * 16 + r + rowhi;
            pd[sbase + (size_t)b * H_DIM + col] = acc[m][r];
        }
    }
}

// -------- kernel 2b: reduce split-K partials -> out (f32) -------------------
__global__ void reduce_out(const float* __restrict__ pd, int ks,
                           float* __restrict__ out, int n) {
    int idx = blockIdx.x * blockDim.x + threadIdx.x;
    if (idx >= n) return;
    float v = 0.0f;
    for (int s = 0; s < ks; ++s) v += pd[(size_t)s * n + idx];
    out[idx] = v;
}

// ---------------------------------------------------------------------------
extern "C" void kernel_launch(void* const* d_in, const int* in_sizes, int n_in,
                              void* d_out, int out_size, void* d_ws, size_t ws_size,
                              hipStream_t stream) {
    const float* x  = (const float*)d_in[0];
    const int*   gq = (const int*)  d_in[1];
    const float* gs = (const float*)d_in[2];
    const float* gz = (const float*)d_in[3];
    const int*   uq = (const int*)  d_in[4];
    const float* us = (const float*)d_in[5];
    const float* uz = (const float*)d_in[6];
    const int*   dq = (const int*)  d_in[7];
    const float* ds = (const float*)d_in[8];
    const float* dz = (const float*)d_in[9];
    float* out = (float*)d_out;

    char* ws = (char*)d_ws;
    _Float16* xh   = (_Float16*)ws;                                // 512 KB
    _Float16* hbuf = (_Float16*)(ws + (size_t)B_DIM * H_DIM * 2);  // 1.38 MB
    const size_t part_off  = 2u * 1024u * 1024u;                   // 2 MB aligned
    const size_t p1_slice  = (size_t)B_DIM * I_DIM * sizeof(float); // 2.82 MB
    const size_t p2_slice  = (size_t)B_DIM * H_DIM * sizeof(float); // 1 MB

    // Largest power-of-two split counts that fit the workspace (deterministic:
    // ws_size is constant across calls).
    int ks1 = 1, ks2 = 1;
    for (int k = 8; k >= 1; k >>= 1)
        if (part_off + 2u * (size_t)k * p1_slice <= ws_size) { ks1 = k; break; }
    for (int k = 8; k >= 1; k >>= 1)
        if (part_off + (size_t)k * p2_slice <= ws_size) { ks2 = k; break; }

    float* pg = (float*)(ws + part_off);
    float* pu = (float*)(ws + part_off + (size_t)ks1 * p1_slice);
    float* pd = (float*)(ws + part_off);   // reused after phase 1 consumed

    const int nx = B_DIM * H_DIM;
    const int nh = B_DIM * I_DIM;

    cvt_x_f16<<<(nx + 255) / 256, 256, 0, stream>>>(x, xh, nx);
    gate_up_partial<<<dim3(I_DIM / 128, ks1), 256, 0, stream>>>(
        xh, gq, gs, gz, uq, us, uz, pg, pu);
    reduce_swiglu<<<(nh + 255) / 256, 256, 0, stream>>>(pg, pu, ks1, hbuf, nh);
    down_partial<<<dim3(H_DIM / 128, ks2), 256, 0, stream>>>(hbuf, dq, ds, dz, pd);
    reduce_out<<<(nx + 255) / 256, 256, 0, stream>>>(pd, ks2, out, nx);
}
